// RangeAwareAttention_45689862094881
// MI455X (gfx1250) — matrix operand order repaired and checked by reading, not compiled, not run on hardware
//
#include <hip/hip_runtime.h>
#include <hip/hip_fp16.h>

typedef __attribute__((ext_vector_type(16))) _Float16 v16h;
typedef __attribute__((ext_vector_type(8)))  _Float16 v8h;
typedef __attribute__((ext_vector_type(8)))  float    v8f;

#define D_MODEL 1024
#define N_HEADS 16
#define D_K     64
#define SEQ     2048
#define BATCH   2
#define ROWS    (BATCH*SEQ)        // 4096
#define QKV_N   (3*D_MODEL)        // 3072
#define GUARD_EPS 0.1f
#define LDK 72                     // padded LDS stride (f16 elems)

// ---------------------------------------------------------------- helpers
__device__ inline v8f wmma16(v16h a, v16h b, v8f c) {
  // D = A(16x32 f16) x B(32x16 f16) + C(16x16 f32)
  return __builtin_amdgcn_wmma_f32_16x16x32_f16(
      false, a, false, b, (short)0, c, false, false);
}

__device__ inline v16h make_v16(v8h lo, v8h hi) {
  return __builtin_shufflevector(lo, hi, 0,1,2,3,4,5,6,7,8,9,10,11,12,13,14,15);
}

__device__ inline float hmax16(float v) {
  #pragma unroll
  for (int m = 1; m < 16; m <<= 1) v = fmaxf(v, __shfl_xor(v, m, 32));
  return v;
}
__device__ inline float hsum16(float v) {
  #pragma unroll
  for (int m = 1; m < 16; m <<= 1) v += __shfl_xor(v, m, 32);
  return v;
}

// CDNA5 async DMA: copy 16B global -> LDS, tracked by ASYNCcnt.
__device__ inline void async_b128(unsigned lds_off, const void* g) {
  asm volatile("global_load_async_to_lds_b128 %0, %1, off"
               :: "v"(lds_off), "v"((unsigned long long)g) : "memory");
}
__device__ inline void async_copy32(unsigned lds_off, const _Float16* g) {
  async_b128(lds_off,      g);
  async_b128(lds_off + 16, (const _Float16*)((const char*)g + 16));
}
__device__ inline unsigned lds_off_of(const void* p) {
  return (unsigned)(unsigned long long)p;   // flat LDS addr: low 32 bits = offset
}
__device__ inline void wait_async0() {
  asm volatile("s_wait_asynccnt 0" ::: "memory");
}

// Pipelined GEMM mainloop: C[32x64] += A[m0..+32,:K] * B[n0..+64,:K]^T
// A,B row-major, ld == K (torch Linear layout). Fragments double-buffered
// in registers so next-step loads overlap current-step WMMAs.
__device__ inline void gemm_tile_32x64(const _Float16* __restrict__ A,
                                       const _Float16* __restrict__ B,
                                       int K, int m0, int n0, int r, int hi,
                                       v8f acc[8]) {
  const _Float16* ar0 = A + (size_t)(m0 + r) * K;
  const _Float16* ar1 = A + (size_t)(m0 + 16 + r) * K;
  const _Float16* br  = B + (size_t)(n0 + r) * K;
  v16h a0 = *(const v16h*)(ar0 + hi*16);
  v16h a1 = *(const v16h*)(ar1 + hi*16);
  v16h b0 = *(const v16h*)(br + (size_t)0*16*K + hi*16);
  v16h b1 = *(const v16h*)(br + (size_t)1*16*K + hi*16);
  v16h b2 = *(const v16h*)(br + (size_t)2*16*K + hi*16);
  v16h b3 = *(const v16h*)(br + (size_t)3*16*K + hi*16);
  for (int k0 = 0; k0 < K; k0 += 32) {
    v16h na0 = a0, na1 = a1, nb0 = b0, nb1 = b1, nb2 = b2, nb3 = b3;
    const int kn = k0 + 32;
    if (kn < K) {   // issue next-step fragment loads before consuming current
      na0 = *(const v16h*)(ar0 + kn + hi*16);
      na1 = *(const v16h*)(ar1 + kn + hi*16);
      nb0 = *(const v16h*)(br + (size_t)0*16*K + kn + hi*16);
      nb1 = *(const v16h*)(br + (size_t)1*16*K + kn + hi*16);
      nb2 = *(const v16h*)(br + (size_t)2*16*K + kn + hi*16);
      nb3 = *(const v16h*)(br + (size_t)3*16*K + kn + hi*16);
    }
    acc[0] = wmma16(a0, b0, acc[0]);
    acc[1] = wmma16(a0, b1, acc[1]);
    acc[2] = wmma16(a0, b2, acc[2]);
    acc[3] = wmma16(a0, b3, acc[3]);
    acc[4] = wmma16(a1, b0, acc[4]);
    acc[5] = wmma16(a1, b1, acc[5]);
    acc[6] = wmma16(a1, b2, acc[6]);
    acc[7] = wmma16(a1, b3, acc[7]);
    a0 = na0; a1 = na1; b0 = nb0; b1 = nb1; b2 = nb2; b3 = nb3;
  }
}

// ---------------------------------------------------------------- kernels
__global__ void cvt_f32_f16(const float* __restrict__ src,
                            _Float16* __restrict__ dst, int n) {
  int i = blockIdx.x * blockDim.x + threadIdx.x;
  int stride = gridDim.x * blockDim.x;
  for (; i < n; i += stride) dst[i] = (_Float16)src[i];
}

// QKV projection + bias + range-guard clamp.
// Q,K scatter to [B,H,S,dk]; V scatters PRE-TRANSPOSED to [B,H,dk,S] so the
// attention kernel can stage V^T tiles with coalesced async copies.
__global__ __launch_bounds__(256) void qkv_gemm(
    const _Float16* __restrict__ X,      // [4096,1024]
    const _Float16* __restrict__ W,      // [3072,1024]
    const float* __restrict__ bias,      // [3072]
    const float* __restrict__ akmin, const float* __restrict__ akmax,
    const float* __restrict__ avmin, const float* __restrict__ avmax,
    _Float16* __restrict__ Qh, _Float16* __restrict__ Kh,
    _Float16* __restrict__ VTh) {
  const int lane = threadIdx.x & 31, wave = threadIdx.x >> 5;
  const int r = lane & 15, hi = lane >> 4;
  const int m0 = blockIdx.x * 256 + wave * 32;
  const int n0 = blockIdx.y * 64;
  v8f acc[8] = {};
  gemm_tile_32x64(X, W, D_MODEL, m0, n0, r, hi, acc);

  #pragma unroll
  for (int mt = 0; mt < 2; ++mt) {
    #pragma unroll
    for (int nt = 0; nt < 4; ++nt) {
      const v8f a = acc[mt*4 + nt];
      const int n = n0 + nt*16 + r;
      const int t = n >> 10;                // 0=q 1=k 2=v
      const int h = (n >> 6) & 15;
      const int c = n & 63;
      const float bn = bias[n];
      #pragma unroll
      for (int i = 0; i < 8; ++i) {
        const int m = m0 + mt*16 + hi*8 + i;
        const int b = m >> 11;
        const int s = m & 2047;
        const int bh = (b << 4) + h;
        float y = a[i] + bn;
        const int ai = bh * D_K + c;
        if (t == 0) {
          Qh[(bh * SEQ + s) * D_K + c] = (_Float16)y;
        } else if (t == 1) {
          float lo = fmaxf(y - GUARD_EPS, akmin[ai]);
          float hh = fminf(y + GUARD_EPS, akmax[ai]);
          lo = fminf(lo, hh);
          y  = fmaxf(lo, fminf(y, hh));
          Kh[(bh * SEQ + s) * D_K + c] = (_Float16)y;
        } else {
          float lo = fmaxf(y - GUARD_EPS, avmin[ai]);
          float hh = fminf(y + GUARD_EPS, avmax[ai]);
          lo = fminf(lo, hh);
          y  = fmaxf(lo, fminf(y, hh));
          VTh[(bh * D_K + c) * SEQ + s] = (_Float16)y;   // transposed store
        }
      }
    }
  }
}

// Flash attention. Block = 8 waves = 128 queries of one (b,h).
// K and V^T tiles double-buffered in LDS via async global->LDS DMA;
// tile t+1 DMA overlaps tile t compute (ASYNCcnt fencing).
__global__ __launch_bounds__(256) void flash_attn(
    const _Float16* __restrict__ Qh, const _Float16* __restrict__ Kh,
    const _Float16* __restrict__ VTh, _Float16* __restrict__ Oh) {
  __shared__ __align__(32) _Float16 Kt[2][64 * LDK];   // [key][dk]
  __shared__ __align__(32) _Float16 Vt[2][64 * LDK];   // [dk][key]
  __shared__ __align__(32) _Float16 Pt[8 * 16 * 32];
  const int tid = threadIdx.x;
  const int lane = tid & 31, wave = tid >> 5;
  const int r = lane & 15, hi = lane >> 4;
  const int bh = blockIdx.y;
  const int b = bh >> 4, h = bh & 15;
  const int m0 = blockIdx.x * 128 + wave * 16;

  const _Float16* Qb = Qh  + (size_t)bh * SEQ * D_K;
  const _Float16* Kb = Kh  + (size_t)bh * SEQ * D_K;
  const _Float16* Vb = VTh + (size_t)bh * D_K * SEQ;

  const _Float16* qrow = Qb + (size_t)(m0 + r) * D_K;
  v16h aq0 = *(const v16h*)(qrow +  0 + hi*16);
  v16h aq1 = *(const v16h*)(qrow + 32 + hi*16);

  float mrow[8], lrow[8];
  v8f o0 = {}, o1 = {}, o2 = {}, o3 = {};
  #pragma unroll
  for (int i = 0; i < 8; ++i) { mrow[i] = -3.0e38f; lrow[i] = 0.f; }

  _Float16* Ptw = Pt + wave * (16 * 32);
  const int srow = tid >> 2;          // 0..63: K key row / V^T channel row
  const int scol = (tid & 3) * 16;    // 16-f16 chunk within row

  auto stage = [&](int kb, int buf) {
    async_copy32(lds_off_of(&Kt[buf][srow * LDK + scol]),
                 Kb + (size_t)(kb + srow) * D_K + scol);
    async_copy32(lds_off_of(&Vt[buf][srow * LDK + scol]),
                 Vb + (size_t)srow * SEQ + kb + scol);
  };

  stage(0, 0);
  wait_async0();
  __syncthreads();

  for (int kb = 0, t = 0; kb < SEQ; kb += 64, ++t) {
    const int cur = t & 1;
    if (kb + 64 < SEQ) stage(kb + 64, cur ^ 1);   // DMA next tile during compute
    const _Float16* Kc = Kt[cur];
    const _Float16* Vc = Vt[cur];

    #pragma unroll
    for (int sub = 0; sub < 2; ++sub) {
      const int kk = sub * 32;
      v8f s0 = {}, s1 = {};
      const _Float16* kb0 = &Kc[(kk + r) * LDK + hi*16];
      const _Float16* kb1 = &Kc[(kk + 16 + r) * LDK + hi*16];
      s0 = wmma16(aq0, *(const v16h*)(kb0),      s0);
      s1 = wmma16(aq0, *(const v16h*)(kb1),      s1);
      s0 = wmma16(aq1, *(const v16h*)(kb0 + 32), s0);
      s1 = wmma16(aq1, *(const v16h*)(kb1 + 32), s1);

      const float scale = 0.125f;     // 1/sqrt(64)
      float p0[8], p1[8], corr[8];
      #pragma unroll
      for (int i = 0; i < 8; ++i) {
        float a  = s0[i] * scale;
        float c2 = s1[i] * scale;
        float tm = hmax16(fmaxf(a, c2));
        float mn = fmaxf(mrow[i], tm);
        corr[i] = __expf(mrow[i] - mn);
        p0[i]   = __expf(a  - mn);
        p1[i]   = __expf(c2 - mn);
        lrow[i] = lrow[i] * corr[i] + hsum16(p0[i] + p1[i]);
        mrow[i] = mn;
      }
      #pragma unroll
      for (int i = 0; i < 8; ++i) {
        o0[i] *= corr[i]; o1[i] *= corr[i];
        o2[i] *= corr[i]; o3[i] *= corr[i];
      }
      // D-layout (lane=key col, elem=query row) -> LDS -> A-layout
      #pragma unroll
      for (int i = 0; i < 8; ++i) {
        Ptw[(hi*8 + i) * 32 + r]      = (_Float16)p0[i];
        Ptw[(hi*8 + i) * 32 + 16 + r] = (_Float16)p1[i];
      }
      asm volatile("s_wait_dscnt 0" ::: "memory");   // wave-local LDS RAW
      v16h pf = *(const v16h*)(&Ptw[r * 32 + hi*16]);
      const _Float16* vbase = &Vc[r * LDK + kk + hi*16];
      o0 = wmma16(pf, *(const v16h*)(vbase + 0*16*LDK), o0);
      o1 = wmma16(pf, *(const v16h*)(vbase + 1*16*LDK), o1);
      o2 = wmma16(pf, *(const v16h*)(vbase + 2*16*LDK), o2);
      o3 = wmma16(pf, *(const v16h*)(vbase + 3*16*LDK), o3);
    }
    wait_async0();       // next tile's DMA landed
    __syncthreads();     // all waves done reading cur before it is recycled
  }

  #pragma unroll
  for (int i = 0; i < 8; ++i) {
    float inv = 1.0f / lrow[i];
    const int s = m0 + hi*8 + i;
    const size_t orow = (size_t)(b * SEQ + s) * D_MODEL + h * D_K;
    Oh[orow + 0*16 + r] = (_Float16)(o0[i] * inv);
    Oh[orow + 1*16 + r] = (_Float16)(o1[i] * inv);
    Oh[orow + 2*16 + r] = (_Float16)(o2[i] * inv);
    Oh[orow + 3*16 + r] = (_Float16)(o3[i] * inv);
  }
}

// Output projection + bias -> f32 d_out.
__global__ __launch_bounds__(256) void out_gemm(
    const _Float16* __restrict__ O,      // [4096,1024]
    const _Float16* __restrict__ W,      // [1024,1024]
    const float* __restrict__ bias,      // [1024]
    float* __restrict__ Y) {             // [4096,1024]
  const int lane = threadIdx.x & 31, wave = threadIdx.x >> 5;
  const int r = lane & 15, hi = lane >> 4;
  const int m0 = blockIdx.x * 256 + wave * 32;
  const int n0 = blockIdx.y * 64;
  v8f acc[8] = {};
  gemm_tile_32x64(O, W, D_MODEL, m0, n0, r, hi, acc);
  #pragma unroll
  for (int mt = 0; mt < 2; ++mt) {
    #pragma unroll
    for (int nt = 0; nt < 4; ++nt) {
      const int n = n0 + nt*16 + r;
      const float bn = bias[n];
      #pragma unroll
      for (int i = 0; i < 8; ++i) {
        const int m = m0 + mt*16 + hi*8 + i;
        Y[(size_t)m * D_MODEL + n] = acc[mt*4 + nt][i] + bn;
      }
    }
  }
}

// ---------------------------------------------------------------- launch
extern "C" void kernel_launch(void* const* d_in, const int* in_sizes, int n_in,
                              void* d_out, int out_size, void* d_ws, size_t ws_size,
                              hipStream_t stream) {
  const float* x     = (const float*)d_in[0];
  const float* qkv_w = (const float*)d_in[1];
  const float* qkv_b = (const float*)d_in[2];
  const float* out_w = (const float*)d_in[3];
  const float* out_b = (const float*)d_in[4];
  const float* akmin = (const float*)d_in[5];
  const float* akmax = (const float*)d_in[6];
  const float* avmin = (const float*)d_in[7];
  const float* avmax = (const float*)d_in[8];
  float* out = (float*)d_out;

  char* ws = (char*)d_ws;
  size_t off = 0;
  auto carve = [&](size_t bytes) {
    void* p = ws + off;
    off = (off + bytes + 255) & ~(size_t)255;
    return p;
  };
  _Float16* XH = (_Float16*)carve((size_t)ROWS * D_MODEL * 2);      // 8 MB
  _Float16* WQ = (_Float16*)carve((size_t)QKV_N * D_MODEL * 2);     // 6 MB
  _Float16* WO = (_Float16*)carve((size_t)D_MODEL * D_MODEL * 2);   // 2 MB
  _Float16* QH = (_Float16*)carve((size_t)ROWS * D_MODEL * 2);      // 8 MB
  _Float16* KH = (_Float16*)carve((size_t)ROWS * D_MODEL * 2);      // 8 MB
  _Float16* VT = (_Float16*)carve((size_t)ROWS * D_MODEL * 2);      // 8 MB
  _Float16* OH = (_Float16*)carve((size_t)ROWS * D_MODEL * 2);      // 8 MB

  cvt_f32_f16<<<1024, 256, 0, stream>>>(x,     XH, ROWS * D_MODEL);
  cvt_f32_f16<<<1024, 256, 0, stream>>>(qkv_w, WQ, QKV_N * D_MODEL);
  cvt_f32_f16<<<512,  256, 0, stream>>>(out_w, WO, D_MODEL * D_MODEL);

  qkv_gemm<<<dim3(ROWS / 256, QKV_N / 64), 256, 0, stream>>>(
      XH, WQ, qkv_b, akmin, akmax, avmin, avmax, QH, KH, VT);

  flash_attn<<<dim3(SEQ / 128, BATCH * N_HEADS), 256, 0, stream>>>(
      QH, KH, VT, OH);

  out_gemm<<<dim3(ROWS / 256, D_MODEL / 64), 256, 0, stream>>>(
      OH, WO, out_b, out);
}
